// AttentionBasedClustering_15058155340285
// MI455X (gfx1250) — matrix operand order repaired and checked by reading, not compiled, not run on hardware
//
#include <hip/hip_runtime.h>
#include <hip/hip_bf16.h>

typedef __attribute__((ext_vector_type(16))) _Float16 v16h;
typedef __attribute__((ext_vector_type(8)))  float    v8f;
typedef __attribute__((ext_vector_type(4)))  unsigned int u32x4;
typedef __attribute__((ext_vector_type(8)))  unsigned int u32x8;

#define WMMA_F32_F16(a, b, c) \
    __builtin_amdgcn_wmma_f32_16x16x32_f16(false, (a), false, (b), (short)0, (c), false, false)

// ---------------------------------------------------------------------------
// Tensor Data Mover: 2D f32 tile, global -> LDS. Two descriptor groups
// (VADDR2/3 omitted => 2D form). Fields per ISA 08 §8.3/§8.4:
//   g0: [1:0]=count=1, [63:32]=lds_addr, [120:64]=global_addr, [127:126]=type=2
//   g1: [17:16]=data_size(2=4B), [79:48]=tensor_dim0, [111:80]=tensor_dim1,
//       [127:112]=tile_dim0, [143:128]=tile_dim1, [207:160]=dim0_stride
// ---------------------------------------------------------------------------
__device__ inline void tdm_load_2d_f32(unsigned int lds_addr, const void* gptr,
                                       unsigned int width, unsigned int height,
                                       unsigned int stride) {
    unsigned long long ga = (unsigned long long)(uintptr_t)gptr;
    u32x4 g0;
    g0[0] = 1u;                                    // count=1, user descriptor
    g0[1] = lds_addr;                              // LDS byte address
    g0[2] = (unsigned int)(ga & 0xFFFFFFFFu);
    g0[3] = (unsigned int)((ga >> 32) & 0x1FFFFFFu) | (2u << 30);  // type=2
    u32x8 g1;
    g1[0] = (2u << 16);                            // data_size = 4 bytes
    g1[1] = (width & 0xFFFFu) << 16;               // tensor_dim0[15:0]
    g1[2] = ((width >> 16) & 0xFFFFu) | ((height & 0xFFFFu) << 16);
    g1[3] = ((height >> 16) & 0xFFFFu) | ((width & 0xFFFFu) << 16);  // tile_dim0
    g1[4] = (height & 0xFFFFu);                    // tile_dim1 (tile_dim2 = 0)
    g1[5] = stride;                                // tensor_dim0_stride[31:0]
    g1[6] = 0u;
    g1[7] = 0u;
    asm volatile("tensor_load_to_lds %0, %1" :: "s"(g0), "s"(g1) : "memory");
}

// ---------------------------------------------------------------------------
// WMMA fragment loaders for 16x16x32 f16, contiguous k order k = 16*(lane/16) + s
// ---------------------------------------------------------------------------
template <int LDA>
__device__ inline v16h load_a_frag_f32(const float* A, int row0, int k0, int lane) {
    int m = lane & 15;
    int kh = (lane >> 4) << 4;
    const float* p = A + (size_t)(row0 + m) * LDA + k0 + kh;
    v16h a;
#pragma unroll
    for (int t = 0; t < 8; ++t) {
        a[t]     = (_Float16)p[t];
        a[8 + t] = (_Float16)p[8 + t];
    }
    return a;
}

template <int LDB, int TRANSB>
__device__ inline v16h load_b_frag_f32(const float* Bm, int k0, int col0, int lane) {
    int n = lane & 15;
    int kh = (lane >> 4) << 4;
    v16h b;
    if (TRANSB) {  // B(k,n) = Bm[n*LDB + k] : contiguous in k
        const float* p = Bm + (size_t)(col0 + n) * LDB + k0 + kh;
#pragma unroll
        for (int t = 0; t < 16; ++t) b[t] = (_Float16)p[t];
    } else {       // B(k,n) = Bm[k*LDB + n] : strided in k (constant stride)
        const float* p = Bm + (size_t)(k0 + kh) * LDB + col0 + n;
#pragma unroll
        for (int t = 0; t < 16; ++t) b[t] = (_Float16)p[t * LDB];
    }
    return b;
}

// ---------------------------------------------------------------------------
// Generic tiled WMMA GEMM: C[M x N] = A[M x K] @ op(B) + bias
// one wave per 16x16 output tile; grid = (N/16, M/16). K/LDB/TRANSB static.
// ---------------------------------------------------------------------------
template <int K, int LDB, int TRANSB>
__global__ __launch_bounds__(32) void gemm_wmma_kernel(
    const float* __restrict__ A, const float* __restrict__ Bm,
    const float* __restrict__ bias, float* __restrict__ C, int N) {
    int lane = threadIdx.x & 31;
    int row0 = blockIdx.y << 4;
    int col0 = blockIdx.x << 4;
    v8f acc = {};
#pragma unroll
    for (int k0 = 0; k0 < K; k0 += 32) {
        v16h a = load_a_frag_f32<K>(A, row0, k0, lane);
        v16h b = load_b_frag_f32<LDB, TRANSB>(Bm, k0, col0, lane);
        acc = WMMA_F32_F16(a, b, acc);
    }
    int n  = lane & 15;
    int mb = (lane >> 4) << 3;
    float bv = bias ? bias[col0 + n] : 0.0f;
#pragma unroll
    for (int v = 0; v < 8; ++v)
        C[(size_t)(row0 + mb + v) * N + col0 + n] = acc[v] + bv;
}

// ---------------------------------------------------------------------------
// Flash-style MHA per (b, h, 16-query tile). One wave per block.
// qkv rows are [q(256) | k(256) | v(256)]; head slice = h*32 + d.
// ---------------------------------------------------------------------------
__global__ __launch_bounds__(32) void attn_kernel(const float* __restrict__ qkv,
                                                  float* __restrict__ O) {
    const int S = 512, RS = 768;
    int lane = threadIdx.x & 31;
    int qt = blockIdx.x, h = blockIdx.y, b = blockIdx.z;
    const float* base = qkv + (size_t)b * S * RS;
    int q0 = qt << 4;

    v16h qa;
    {
        int m = lane & 15, kh = (lane >> 4) << 4;
        const float* p = base + (size_t)(q0 + m) * RS + h * 32 + kh;
        const float scale = 0.17677669529663687f;  // 1/sqrt(32)
#pragma unroll
        for (int t = 0; t < 8; ++t) {
            qa[t]     = (_Float16)(p[t] * scale);
            qa[8 + t] = (_Float16)(p[8 + t] * scale);
        }
    }

    float mrow[8], lrow[8];
    v8f o0 = {}, o1 = {};
#pragma unroll
    for (int v = 0; v < 8; ++v) { mrow[v] = -1e30f; lrow[v] = 0.0f; }

    __shared__ _Float16 pbuf[16 * 32];

    for (int jc = 0; jc < S; jc += 32) {
        v8f s0, s1;
#pragma unroll
        for (int half = 0; half < 2; ++half) {
            int n = lane & 15, kh = (lane >> 4) << 4;
            const float* kp = base + (size_t)(jc + half * 16 + n) * RS + 256 + h * 32 + kh;
            v16h kb;
#pragma unroll
            for (int t = 0; t < 16; ++t) kb[t] = (_Float16)kp[t];
            v8f z = {};
            if (half == 0) s0 = WMMA_F32_F16(qa, kb, z);
            else           s1 = WMMA_F32_F16(qa, kb, z);
        }
        float p0[8], p1[8];
#pragma unroll
        for (int v = 0; v < 8; ++v) {
            float mx = fmaxf(s0[v], s1[v]);
#pragma unroll
            for (int off = 1; off < 16; off <<= 1)
                mx = fmaxf(mx, __shfl_xor(mx, off, 16));
            float mnew = fmaxf(mrow[v], mx);
            p0[v] = __expf(s0[v] - mnew);
            p1[v] = __expf(s1[v] - mnew);
            float ls = p0[v] + p1[v];
#pragma unroll
            for (int off = 1; off < 16; off <<= 1) ls += __shfl_xor(ls, off, 16);
            float corr = __expf(mrow[v] - mnew);
            lrow[v] = lrow[v] * corr + ls;
            o0[v] *= corr;
            o1[v] *= corr;
            mrow[v] = mnew;
        }
        {
            int col = lane & 15, rb = (lane >> 4) << 3;
#pragma unroll
            for (int v = 0; v < 8; ++v) {
                pbuf[(rb + v) * 32 + col]      = (_Float16)p0[v];
                pbuf[(rb + v) * 32 + 16 + col] = (_Float16)p1[v];
            }
        }
        __syncthreads();
        v16h pa;
        {
            int m = lane & 15, kh = (lane >> 4) << 4;
            const _Float16* pp = pbuf + m * 32 + kh;
#pragma unroll
            for (int t = 0; t < 8; ++t) { pa[t] = pp[t]; pa[8 + t] = pp[8 + t]; }
        }
#pragma unroll
        for (int nt = 0; nt < 2; ++nt) {
            int n = lane & 15, kh = (lane >> 4) << 4;
            const float* vp = base + (size_t)(jc + kh) * RS + 512 + h * 32 + nt * 16 + n;
            v16h vb;
#pragma unroll
            for (int t = 0; t < 16; ++t) vb[t] = (_Float16)vp[(size_t)t * RS];
            if (nt == 0) o0 = WMMA_F32_F16(pa, vb, o0);
            else         o1 = WMMA_F32_F16(pa, vb, o1);
        }
        __syncthreads();
    }
    int n = lane & 15, mb = (lane >> 4) << 3;
#pragma unroll
    for (int v = 0; v < 8; ++v) {
        float inv = 1.0f / lrow[v];
        size_t r = ((size_t)b * S + q0 + mb + v) * 256 + h * 32;
        O[r + n]      = o0[v] * inv;
        O[r + 16 + n] = o1[v] * inv;
    }
}

// ---------------------------------------------------------------------------
// x_out = LayerNorm(x_in + y) * g + b ; one wave per 256-wide row
// ---------------------------------------------------------------------------
__global__ __launch_bounds__(256) void residual_ln_kernel(
    const float* __restrict__ xin, const float* __restrict__ y,
    const float* __restrict__ g, const float* __restrict__ bb,
    float* __restrict__ xout) {
    int wave = threadIdx.x >> 5, lane = threadIdx.x & 31;
    int row = (blockIdx.x << 3) + wave;
    const float* xr = xin + (size_t)row * 256;
    const float* yr = y + (size_t)row * 256;
    float vals[8], s = 0.0f;
#pragma unroll
    for (int t = 0; t < 8; ++t) {
        vals[t] = xr[lane + 32 * t] + yr[lane + 32 * t];
        s += vals[t];
    }
#pragma unroll
    for (int off = 1; off < 32; off <<= 1) s += __shfl_xor(s, off, 32);
    float mean = s * (1.0f / 256.0f);
    float vs = 0.0f;
#pragma unroll
    for (int t = 0; t < 8; ++t) { float d = vals[t] - mean; vs += d * d; }
#pragma unroll
    for (int off = 1; off < 32; off <<= 1) vs += __shfl_xor(vs, off, 32);
    float r = rsqrtf(vs * (1.0f / 256.0f) + 1e-5f);
#pragma unroll
    for (int t = 0; t < 8; ++t) {
        int c = lane + 32 * t;
        xout[(size_t)row * 256 + c] = (vals[t] - mean) * r * g[c] + bb[c];
    }
}

// ---------------------------------------------------------------------------
// Fused pairwise head. Block = 8 waves, one i per block, 128 j's per block,
// 16 j-pairs per wave (= WMMA M dim). W2 staged via Tensor Data Mover.
// grid = (S/128, S, B)
// ---------------------------------------------------------------------------
__global__ __launch_bounds__(256) void pairwise_kernel(
    const float* __restrict__ ai, const float* __restrict__ bj,
    const float* __restrict__ W2, const float* __restrict__ b1,
    const float* __restrict__ b2, const float* __restrict__ W3,
    const float* __restrict__ b3, float* __restrict__ out) {
    __shared__ float    w2f[128 * 64];   // TDM landing buffer (f32)
    __shared__ _Float16 w2s[128 * 64];   // converted f16 operand tile
    __shared__ float aib[128];
    __shared__ float w3s[64];
    __shared__ float b2s[64];

    int tid = threadIdx.x;
    int jblk = blockIdx.x, i = blockIdx.y, b = blockIdx.z;

    // --- TDM: DMA the whole 128x64 f32 W2 tile into LDS (wave 0 issues) ---
    if (tid < 32) {
        tdm_load_2d_f32((unsigned int)(uintptr_t)w2f, W2, 64u, 128u, 64u);
        __builtin_amdgcn_s_wait_tensorcnt(0);
    }
    const float* air = ai + ((size_t)b * 512 + i) * 128;
    if (tid < 128) aib[tid] = air[tid] + b1[tid];
    if (tid >= 128 && tid < 192) { w3s[tid - 128] = W3[tid - 128]; b2s[tid - 128] = b2[tid - 128]; }
    __syncthreads();
    // --- convert W2 to f16 in LDS ---
    for (int idx = tid; idx < 128 * 64; idx += 256) w2s[idx] = (_Float16)w2f[idx];
    __syncthreads();

    int wave = tid >> 5, lane = tid & 31;
    int j0 = (jblk << 7) + (wave << 4);
    const float* bjb = bj + (size_t)b * 512 * 128;

    v8f acc0 = {}, acc1 = {}, acc2 = {}, acc3 = {};
#pragma unroll
    for (int k0 = 0; k0 < 128; k0 += 32) {
        // A fragment: h1 tile = relu(ai[i] + bj[j0+m] + b1), formed on the fly
        v16h af;
        {
            int m = lane & 15, kh = (lane >> 4) << 4;
            const float* bp = bjb + (size_t)(j0 + m) * 128 + k0 + kh;
            __builtin_prefetch(bp + 32, 0, 0);
#pragma unroll
            for (int t = 0; t < 8; ++t) {
                af[t]     = (_Float16)fmaxf(aib[k0 + kh + t] + bp[t], 0.0f);
                af[8 + t] = (_Float16)fmaxf(aib[k0 + 8 + kh + t] + bp[8 + t], 0.0f);
            }
        }
        int n = lane & 15, kh = (lane >> 4) << 4;
#pragma unroll
        for (int nt = 0; nt < 4; ++nt) {
            const _Float16* wp = w2s + (size_t)(k0 + kh) * 64 + nt * 16 + n;
            v16h bf;
#pragma unroll
            for (int t = 0; t < 16; ++t) bf[t] = wp[t * 64];
            if (nt == 0)      acc0 = WMMA_F32_F16(af, bf, acc0);
            else if (nt == 1) acc1 = WMMA_F32_F16(af, bf, acc1);
            else if (nt == 2) acc2 = WMMA_F32_F16(af, bf, acc2);
            else              acc3 = WMMA_F32_F16(af, bf, acc3);
        }
    }
    // h2 = relu(acc + b2); logit = h2 . W3 ; reduce across 16 lanes per row
    int n = lane & 15;
    float part[8];
#pragma unroll
    for (int v = 0; v < 8; ++v) {
        float p = 0.0f;
        p += fmaxf(acc0[v] + b2s[n],      0.0f) * w3s[n];
        p += fmaxf(acc1[v] + b2s[16 + n], 0.0f) * w3s[16 + n];
        p += fmaxf(acc2[v] + b2s[32 + n], 0.0f) * w3s[32 + n];
        p += fmaxf(acc3[v] + b2s[48 + n], 0.0f) * w3s[48 + n];
        part[v] = p;
    }
#pragma unroll
    for (int v = 0; v < 8; ++v)
#pragma unroll
        for (int off = 1; off < 16; off <<= 1)
            part[v] += __shfl_xor(part[v], off, 16);
    if ((lane & 15) == 0) {
        int mb = (lane >> 4) << 3;
        float bb3 = b3[0];
#pragma unroll
        for (int v = 0; v < 8; ++v) {
            float lg = part[v] + bb3;
            out[((size_t)b * 512 + i) * 512 + j0 + mb + v] = 1.0f / (1.0f + __expf(-lg));
        }
    }
}

// ---------------------------------------------------------------------------
extern "C" void kernel_launch(void* const* d_in, const int* in_sizes, int n_in,
                              void* d_out, int out_size, void* d_ws, size_t ws_size,
                              hipStream_t stream) {
    (void)in_sizes; (void)n_in; (void)out_size; (void)ws_size;
    const int D = 256, S = 512, B = 4, Rows = B * S;  // 2048

    const float* emb        = (const float*)d_in[0];
    const float* in_proj_w  = (const float*)d_in[1];   // (2, 768, 256)
    const float* in_proj_b  = (const float*)d_in[2];   // (2, 768)
    const float* out_proj_w = (const float*)d_in[3];   // (2, 256, 256)
    const float* out_proj_b = (const float*)d_in[4];   // (2, 256)
    const float* ln_g       = (const float*)d_in[5];   // (2, 256)
    const float* ln_b       = (const float*)d_in[6];   // (2, 256)
    const float* W1         = (const float*)d_in[7];   // (512, 128)
    const float* b1         = (const float*)d_in[8];   // (128,)
    const float* W2         = (const float*)d_in[9];   // (128, 64)
    const float* b2         = (const float*)d_in[10];  // (64,)
    const float* W3         = (const float*)d_in[11];  // (64, 1)
    const float* b3         = (const float*)d_in[12];  // (1,)
    float* out = (float*)d_out;

    float* ws  = (float*)d_ws;
    float* qkv = ws;                       // 2048*768
    float* o   = qkv + (size_t)Rows * 768; // 2048*256
    float* y   = o + (size_t)Rows * D;     // 2048*256
    float* x1  = y + (size_t)Rows * D;     // 2048*256
    float* x2  = x1 + (size_t)Rows * D;    // 2048*256
    float* aiw = x2 + (size_t)Rows * D;    // 2048*128
    float* bjw = aiw + (size_t)Rows * 128; // 2048*128

    dim3 gQKV(768 / 16, Rows / 16);
    dim3 gOUT(D / 16, Rows / 16);
    dim3 gAB(128 / 16, Rows / 16);
    dim3 gATT(S / 16, 8, B);
    dim3 gLN(Rows / 8);
    dim3 gPW(S / 128, S, B);

    const float* xin = emb;
    float* xouts[2] = {x1, x2};
    for (int l = 0; l < 2; ++l) {
        gemm_wmma_kernel<256, 256, 1><<<gQKV, 32, 0, stream>>>(
            xin, in_proj_w + (size_t)l * 768 * D, in_proj_b + (size_t)l * 768,
            qkv, 768);
        attn_kernel<<<gATT, 32, 0, stream>>>(qkv, o);
        gemm_wmma_kernel<256, 256, 1><<<gOUT, 32, 0, stream>>>(
            o, out_proj_w + (size_t)l * D * D, out_proj_b + (size_t)l * D,
            y, D);
        residual_ln_kernel<<<gLN, 256, 0, stream>>>(
            xin, y, ln_g + (size_t)l * D, ln_b + (size_t)l * D, xouts[l]);
        xin = xouts[l];
    }

    // ai = x @ W1[:256], bj = x @ W1[256:]   (b1 folded into pairwise kernel)
    gemm_wmma_kernel<256, 128, 0><<<gAB, 32, 0, stream>>>(x2, W1, nullptr, aiw, 128);
    gemm_wmma_kernel<256, 128, 0><<<gAB, 32, 0, stream>>>(x2, W1 + (size_t)D * 128,
                                                          nullptr, bjw, 128);

    pairwise_kernel<<<gPW, 256, 0, stream>>>(aiw, bjw, W2, b1, b2, W3, b3, out);
}
